// SelfAttention_7026566496295
// MI455X (gfx1250) — matrix operand order repaired and checked by reading, not compiled, not run on hardware
//
#include <hip/hip_runtime.h>

// ---------------- problem constants ----------------
#define C_DIM   128
#define NPIX    4096          // 64*64
#define NBATCH  8
#define CPG     4             // channels per group (128/32)
#define QSCALE  0.08838834764831845f   // 1/sqrt(128)

typedef _Float16 half_t;
typedef __attribute__((ext_vector_type(16))) _Float16 v16h;
typedef __attribute__((ext_vector_type(8)))  _Float16 v8h;
typedef __attribute__((ext_vector_type(8)))  float    v8f;

union V16U { v16h v; v8h h[2]; };

// B-matrix fragment: 16 contiguous halves (lane<16 -> K 0..15, lane>=16 -> K 16..31;
// caller adds sel*16 to the pointer).
static __device__ __forceinline__ v16h load_b_frag(const half_t* p) {
  V16U u;
  u.h[0] = *(const v8h*)(p);
  u.h[1] = *(const v8h*)(p + 8);
  return u.v;
}
// A-matrix fragment: lane<16 holds K 0..15, lane>=16 holds K 16..31
// rowk points at the start of this lane's row's 32-length K chunk; sel = lane>>4.
static __device__ __forceinline__ v16h load_a_frag(const half_t* rowk, int sel) {
  V16U u;
  u.h[0] = *(const v8h*)(rowk + sel * 16);
  u.h[1] = *(const v8h*)(rowk + sel * 16 + 8);
  return u.v;
}
static __device__ __forceinline__ v8f wmma_f16(v16h a, v16h b, v8f c) {
  return __builtin_amdgcn_wmma_f32_16x16x32_f16(false, a, false, b, (short)0, c,
                                                false, false);
}

// CDNA5 async DMA: copy 16 bytes global -> LDS without touching VGPR data path.
// Tracked by ASYNCcnt; lds_off is the raw LDS byte address (= low 32 bits of the
// generic pointer per the flat-aperture mapping).
static __device__ __forceinline__ void async_ld16(unsigned lds_off, const half_t* g) {
  asm volatile("global_load_async_to_lds_b128 %0, %1, off"
               :: "v"(lds_off), "v"(g)
               : "memory");
}
static __device__ __forceinline__ void wait_async0() {
  asm volatile("s_wait_asynccnt 0x0" ::: "memory");
}

// ---------------- kernel 1: GroupNorm -> hT [B][n][C] f16 ----------------
__global__ void __launch_bounds__(256) gn_kernel(const float* __restrict__ x,
                                                 const float* __restrict__ gamma,
                                                 const float* __restrict__ beta,
                                                 half_t* __restrict__ hT) {
  const int g = blockIdx.x;         // 0..31
  const int b = blockIdx.y;         // 0..7
  const int t = threadIdx.x;
  const float* xg = x + ((size_t)b * C_DIM + (size_t)g * CPG) * NPIX;

  __shared__ float s_sum[256];
  __shared__ float s_sq[256];
  float sum = 0.f, sq = 0.f;
  for (int idx = t; idx < CPG * NPIX; idx += 256) {
    float v = xg[idx];
    sum += v;
    sq  += v * v;
  }
  s_sum[t] = sum; s_sq[t] = sq;
  __syncthreads();
  for (int off = 128; off > 0; off >>= 1) {
    if (t < off) { s_sum[t] += s_sum[t + off]; s_sq[t] += s_sq[t + off]; }
    __syncthreads();
  }
  const float inv_n = 1.f / (float)(CPG * NPIX);
  const float mean  = s_sum[0] * inv_n;
  const float var   = s_sq[0] * inv_n - mean * mean;
  const float rinv  = rsqrtf(var + 1e-6f);

  for (int idx = t; idx < CPG * NPIX; idx += 256) {
    int cc = idx >> 12;              // /4096
    int i  = idx & (NPIX - 1);
    int c  = g * CPG + cc;
    float v = (xg[idx] - mean) * rinv * gamma[c] + beta[c];
    hT[((size_t)b * NPIX + i) * C_DIM + c] = (half_t)v;
  }
}

// ---------------- kernel 2: weights f32 -> f16 ----------------
__global__ void __launch_bounds__(256) wcvt_kernel(const float* __restrict__ wq,
                                                   const float* __restrict__ wk,
                                                   const float* __restrict__ wv,
                                                   const float* __restrict__ wo,
                                                   half_t* __restrict__ whq,
                                                   half_t* __restrict__ whk,
                                                   half_t* __restrict__ whv,
                                                   half_t* __restrict__ who) {
  int idx = blockIdx.x * 256 + threadIdx.x;   // 0..65535
  int sel = idx >> 14;
  int r   = idx & 16383;
  const float* src = (sel == 0) ? wq : (sel == 1) ? wk : (sel == 2) ? wv : wo;
  half_t*      dst = (sel == 0) ? whq : (sel == 1) ? whk : (sel == 2) ? whv : who;
  dst[r] = (half_t)src[r];
}

// ---------------- kernel 3: QKV projections via WMMA ----------------
// qT,kT: [B][n][C] f16 (q pre-scaled by 1/sqrt(C));  v: [B][C][n] f16
__global__ void __launch_bounds__(256) qkv_kernel(const half_t* __restrict__ hT,
                                                  const half_t* __restrict__ whq,
                                                  const half_t* __restrict__ whk,
                                                  const half_t* __restrict__ whv,
                                                  const float* __restrict__ bq,
                                                  const float* __restrict__ bk,
                                                  const float* __restrict__ bv,
                                                  half_t* __restrict__ qT,
                                                  half_t* __restrict__ kT,
                                                  half_t* __restrict__ v) {
  const int b    = blockIdx.y;
  const int i0   = blockIdx.x * 128;
  const int wave = threadIdx.x >> 5;
  const int lane = threadIdx.x & 31;
  const int lo   = lane & 15;
  const int sel  = lane >> 4;
  const int iw   = i0 + wave * 16;      // this wave's 16 pixels

  const half_t* hTb = hT + (size_t)b * NPIX * C_DIM;

  // ---- Q and K: D(M=pixel, N=cout), A = hT rows ----
  v16h afr[4];
  {
    const half_t* arow = hTb + (size_t)(iw + lo) * C_DIM;
#pragma unroll
    for (int kc = 0; kc < 4; kc++) afr[kc] = load_a_frag(arow + kc * 32, sel);
  }
#pragma unroll
  for (int pass = 0; pass < 2; pass++) {
    const half_t* wh  = pass ? whk : whq;
    const float* bias = pass ? bk : bq;
    half_t* outp      = pass ? kT : qT;
    const float scl   = pass ? 1.0f : QSCALE;
    half_t* ob = outp + (size_t)b * NPIX * C_DIM;
#pragma unroll
    for (int ct = 0; ct < 8; ct++) {
      const int cout = ct * 16 + lo;
      v8f acc = {};
#pragma unroll
      for (int kc = 0; kc < 4; kc++) {
        v16h bfr = load_b_frag(wh + (size_t)cout * C_DIM + kc * 32 + sel * 16);
        acc = wmma_f16(afr[kc], bfr, acc);
      }
      const float bb = bias[cout];
#pragma unroll
      for (int r = 0; r < 8; r++) {
        int i = iw + r + 8 * sel;
        ob[(size_t)i * C_DIM + cout] = (half_t)((acc[r] + bb) * scl);
      }
    }
  }

  // ---- V: D(M=cout, N=pixel), A = wv rows, B = hT rows (per-lane pixel) ----
  v16h bfr2[4];
#pragma unroll
  for (int kc = 0; kc < 4; kc++)
    bfr2[kc] = load_b_frag(hTb + (size_t)(iw + lo) * C_DIM + kc * 32 + sel * 16);

  half_t* vb = v + (size_t)b * C_DIM * NPIX;
#pragma unroll
  for (int mt = 0; mt < 8; mt++) {
    v8f acc = {};
#pragma unroll
    for (int kc = 0; kc < 4; kc++) {
      v16h af = load_a_frag(whv + (size_t)(mt * 16 + lo) * C_DIM + kc * 32, sel);
      acc = wmma_f16(af, bfr2[kc], acc);
    }
#pragma unroll
    for (int r = 0; r < 8; r++) {
      int cout = mt * 16 + r + 8 * sel;
      vb[(size_t)cout * NPIX + iw + lo] = (half_t)(acc[r] + bv[cout]);
    }
  }
}

// ---------------- kernel 4: fused flash attention + proj_out + residual ----
#define BN        64
#define KT_STRIDE 136   // halves (padded: 128 + 8) -> 272B rows, 16B-aligned
#define V_STRIDE  72    // halves (padded: 64 + 8)  -> 144B rows, 16B-aligned
#define P_STRIDE  72
#define KT_BYTES  (64 * KT_STRIDE * 2)        // 17408
#define V_OFF     KT_BYTES
#define V_BYTES   (128 * V_STRIDE * 2)        // 18432
#define P_OFF     (V_OFF + V_BYTES)           // 35840
#define P_BYTES   (8 * 16 * P_STRIDE * 2)     // 18432
#define SMEM_SZ   (P_OFF + P_BYTES)           // 54272

__global__ void __launch_bounds__(256) attn_kernel(const half_t* __restrict__ qT,
                                                   const half_t* __restrict__ kTg,
                                                   const half_t* __restrict__ vg,
                                                   const half_t* __restrict__ who,
                                                   const float* __restrict__ bo,
                                                   const float* __restrict__ x,
                                                   float* __restrict__ out) {
  __shared__ __align__(16) unsigned char smem[SMEM_SZ];
  half_t* kTl = (half_t*)smem;                 // [64][KT_STRIDE]
  half_t* vl  = (half_t*)(smem + V_OFF);       // [128][V_STRIDE]
  half_t* Pl  = (half_t*)(smem + P_OFF);       // [8 waves][16][P_STRIDE]

  const int b    = blockIdx.y;
  const int i0   = blockIdx.x * 128;
  const int tid  = threadIdx.x;
  const int wave = tid >> 5;
  const int lane = tid & 31;
  const int lo   = lane & 15;
  const int sel  = lane >> 4;
  const int iw   = i0 + wave * 16;

  const half_t* qTb = qT  + (size_t)b * NPIX * C_DIM;
  const half_t* kTb = kTg + (size_t)b * NPIX * C_DIM;
  const half_t* vb  = vg  + (size_t)b * C_DIM * NPIX;

  // Raw LDS byte addresses for the async DMA destination operand.
  const unsigned kt_base = (unsigned)(uintptr_t)kTl;
  const unsigned v_base  = (unsigned)(uintptr_t)vl;

  // Q A-fragments held in registers for the whole kernel (K = 128 -> 4 chunks)
  v16h qfr[4];
  {
    const half_t* arow = qTb + (size_t)(iw + lo) * C_DIM;
#pragma unroll
    for (int kc = 0; kc < 4; kc++) qfr[kc] = load_a_frag(arow + kc * 32, sel);
  }

  v8f o[8];
#pragma unroll
  for (int ct = 0; ct < 8; ct++) o[ct] = {};
  float m[8], lsum[8];
#pragma unroll
  for (int r = 0; r < 8; r++) { m[r] = -1e30f; lsum[r] = 0.f; }

  half_t* Pw = Pl + wave * 16 * P_STRIDE;

  for (int j0 = 0; j0 < NPIX; j0 += BN) {
    __syncthreads();   // previous iteration's LDS readers are done

    // ---- stage K^T [64][128] and V [128][64] tiles via async DMA to LDS ----
    // K tile: 1024 x 16B chunks, 8 chunks per row of 128 halves.
#pragma unroll
    for (int c4 = 0; c4 < 4; c4++) {
      int u  = tid + c4 * 256;
      int jr = u >> 3, cu = u & 7;
      async_ld16(kt_base + jr * (KT_STRIDE * 2) + cu * 16,
                 kTb + (size_t)(j0 + jr) * C_DIM + cu * 8);
    }
    // V tile: 1024 x 16B chunks, 4 chunks per row of 64 halves.
#pragma unroll
    for (int c4 = 0; c4 < 4; c4++) {
      int u  = tid + c4 * 256;
      int cr = u >> 2, ju = u & 3;
      async_ld16(v_base + cr * (V_STRIDE * 2) + ju * 16,
                 vb + (size_t)cr * NPIX + j0 + ju * 8);
    }
    wait_async0();     // this wave's DMA chunks have landed in LDS
    __syncthreads();   // ... and everyone else's are visible too

    if (j0 + BN < NPIX) {  // uniform branch: L2 prefetch of next K tile
      __builtin_prefetch((const char*)(kTb + (size_t)(j0 + BN) * C_DIM) + tid * 64, 0, 1);
    }

    // S = Q^T K  (logits already scaled; q was pre-scaled by 1/sqrt(C))
    v8f s[4];
#pragma unroll
    for (int jt = 0; jt < 4; jt++) {
      v8f acc = {};
      const half_t* brow = kTl + (jt * 16 + lo) * KT_STRIDE + sel * 16;
#pragma unroll
      for (int kc = 0; kc < 4; kc++)
        acc = wmma_f16(qfr[kc], load_b_frag(brow + kc * 32), acc);
      s[jt] = acc;
    }

    // online softmax (rows live in 16-lane halves of the wave)
    float mnew[8];
#pragma unroll
    for (int r = 0; r < 8; r++) {
      float mx = fmaxf(fmaxf(s[0][r], s[1][r]), fmaxf(s[2][r], s[3][r]));
#pragma unroll
      for (int d = 1; d < 16; d <<= 1) mx = fmaxf(mx, __shfl_xor(mx, d, 32));
      mnew[r] = fmaxf(m[r], mx);
      float corr = __expf(m[r] - mnew[r]);
      lsum[r] *= corr;
#pragma unroll
      for (int ct = 0; ct < 8; ct++) o[ct][r] *= corr;
      float psum = 0.f;
#pragma unroll
      for (int jt = 0; jt < 4; jt++) {
        float p = __expf(s[jt][r] - mnew[r]);
        s[jt][r] = p;
        psum += p;
      }
#pragma unroll
      for (int d = 1; d < 16; d <<= 1) psum += __shfl_xor(psum, d, 32);
      lsum[r] += psum;
      m[r] = mnew[r];
    }

    // transpose P (D-layout) -> A-layout via per-wave LDS scratch
#pragma unroll
    for (int r = 0; r < 8; r++)
#pragma unroll
      for (int jt = 0; jt < 4; jt++)
        Pw[(r + 8 * sel) * P_STRIDE + jt * 16 + lo] = (half_t)s[jt][r];

    // O += P * V^T   (K over keys: 2 chunks of 32)
#pragma unroll
    for (int kc2 = 0; kc2 < 2; kc2++) {
      v16h pfr = load_a_frag(Pw + lo * P_STRIDE + kc2 * 32, sel);
#pragma unroll
      for (int ct = 0; ct < 8; ct++) {
        const half_t* brow = vl + (ct * 16 + lo) * V_STRIDE + kc2 * 32 + sel * 16;
        o[ct] = wmma_f16(pfr, load_b_frag(brow), o[ct]);
      }
    }
  }

  __syncthreads();   // done with K/V tiles; reuse that LDS for O^T

  // normalize by row sums and write O^T (f16) to per-wave LDS [16][KT_STRIDE]
  half_t* OTl = (half_t*)(smem + wave * (16 * KT_STRIDE * 2));
#pragma unroll
  for (int r = 0; r < 8; r++) {
    float inv = 1.f / lsum[r];
#pragma unroll
    for (int ct = 0; ct < 8; ct++)
      OTl[(r + 8 * sel) * KT_STRIDE + ct * 16 + lo] = (half_t)(o[ct][r] * inv);
  }

  // proj_out: D(M=cout, N=query), A = wo rows (global f16), B = O^T rows (LDS)
  v16h bfo[4];
#pragma unroll
  for (int kc = 0; kc < 4; kc++)
    bfo[kc] = load_b_frag(OTl + lo * KT_STRIDE + kc * 32 + sel * 16);

  const float* xb = x + (size_t)b * C_DIM * NPIX;
  float* ob = out + (size_t)b * C_DIM * NPIX;
#pragma unroll
  for (int mt = 0; mt < 8; mt++) {
    v8f acc = {};
#pragma unroll
    for (int kc = 0; kc < 4; kc++) {
      v16h af = load_a_frag(who + (size_t)(mt * 16 + lo) * C_DIM + kc * 32, sel);
      acc = wmma_f16(af, bfo[kc], acc);
    }
#pragma unroll
    for (int r = 0; r < 8; r++) {
      int cout = mt * 16 + r + 8 * sel;
      size_t idx = (size_t)cout * NPIX + iw + lo;   // coalesced across lanes
      ob[idx] = xb[idx] + acc[r] + bo[cout];
    }
  }
}

// ---------------- host launcher ----------------
extern "C" void kernel_launch(void* const* d_in, const int* in_sizes, int n_in,
                              void* d_out, int out_size, void* d_ws, size_t ws_size,
                              hipStream_t stream) {
  const float* x     = (const float*)d_in[0];
  const float* gamma = (const float*)d_in[1];
  const float* beta  = (const float*)d_in[2];
  const float* wq    = (const float*)d_in[3];
  const float* bq    = (const float*)d_in[4];
  const float* wk    = (const float*)d_in[5];
  const float* bk    = (const float*)d_in[6];
  const float* wv    = (const float*)d_in[7];
  const float* bv    = (const float*)d_in[8];
  const float* wo    = (const float*)d_in[9];
  const float* bo    = (const float*)d_in[10];
  float* out = (float*)d_out;

  unsigned char* ws = (unsigned char*)d_ws;
  const size_t ACT = (size_t)NBATCH * NPIX * C_DIM * sizeof(half_t);  // 8 MB
  half_t* hT  = (half_t*)(ws);
  half_t* qT  = (half_t*)(ws + ACT);
  half_t* kT  = (half_t*)(ws + 2 * ACT);
  half_t* v   = (half_t*)(ws + 3 * ACT);
  half_t* whq = (half_t*)(ws + 4 * ACT);
  half_t* whk = whq + C_DIM * C_DIM;
  half_t* whv = whk + C_DIM * C_DIM;
  half_t* who = whv + C_DIM * C_DIM;

  gn_kernel<<<dim3(32, NBATCH), 256, 0, stream>>>(x, gamma, beta, hT);
  wcvt_kernel<<<256, 256, 0, stream>>>(wq, wk, wv, wo, whq, whk, whv, who);
  qkv_kernel<<<dim3(NPIX / 128, NBATCH), 256, 0, stream>>>(
      hT, whq, whk, whv, bq, bk, bv, qT, kT, v);
  attn_kernel<<<dim3(NPIX / 128, NBATCH), 256, 0, stream>>>(
      qT, kT, v, who, bo, x, out);
}
